// NNUE_81080392614274
// MI455X (gfx1250) — compile-verified, matrix operand order checked by reading, not run on hardware
//
#include <hip/hip_runtime.h>
#include <hip/hip_bf16.h>

// ---------------------------------------------------------------------------
// NNUE forward for MI455X (gfx1250, wave32, WMMA).
//  Heavy op: stacked-features [16384 x 10240] @ ft_w.T [10240 x 256] in f16
//  WMMA (v_wmma_f32_16x16x32_f16), f32 accumulate. Small MLP fused per-wave.
// ---------------------------------------------------------------------------

typedef __attribute__((ext_vector_type(16))) _Float16 v16h;
typedef __attribute__((ext_vector_type(8)))  _Float16 v8h;
typedef __attribute__((ext_vector_type(8)))  float    v8f;
typedef __attribute__((ext_vector_type(4)))  float    v4f;

#define N_INPUTS 10240
#define N_H0     256
#define BATCH    8192

__device__ __forceinline__ v8f wmma_f16(v16h a, v16h b, v8f c) {
  return __builtin_amdgcn_wmma_f32_16x16x32_f16(
      /*neg_a=*/false, a, /*neg_b=*/false, b,
      /*c_mod=*/(short)0, c, /*reuse_a=*/false, /*reuse_b=*/false);
}

// Per-lane 16-bit operand fragment, ISA layout for 16x32 A (and symmetric B):
// lanes 0-15 hold K = {0..7, 16..23}+k0, lanes 16-31 hold K = {8..15, 24..31}+k0
// (caller pre-biases pointer by +8 elements for lanes 16-31).
__device__ __forceinline__ v16h frag_from_halfs(const _Float16* p) {
  v8h lo = *(const v8h*)(p);
  v8h hi = *(const v8h*)(p + 16);
  v16h r;
#pragma unroll
  for (int i = 0; i < 8; ++i) { r[i] = lo[i]; r[i + 8] = hi[i]; }
  return r;
}

__device__ __forceinline__ v16h frag_from_f32(const float* p) {
  v4f x0 = *(const v4f*)(p);
  v4f x1 = *(const v4f*)(p + 4);
  v4f x2 = *(const v4f*)(p + 16);
  v4f x3 = *(const v4f*)(p + 20);
  v16h r;
#pragma unroll
  for (int i = 0; i < 4; ++i) {
    r[i]      = (_Float16)x0[i];
    r[4 + i]  = (_Float16)x1[i];
    r[8 + i]  = (_Float16)x2[i];
    r[12 + i] = (_Float16)x3[i];
  }
  return r;
}

__device__ __forceinline__ float clamp01(float x) {
  return x < 0.f ? 0.f : (x > 1.f ? 1.f : x);
}

// ---------------------------------------------------------------------------
// Kernel 0: convert weights f32 -> f16 into workspace.
// ---------------------------------------------------------------------------
__global__ void nnue_cvt(const float* __restrict__ ft_w,
                         const float* __restrict__ l1_w,
                         const float* __restrict__ l2_w,
                         _Float16* __restrict__ ft16,
                         _Float16* __restrict__ l116,
                         _Float16* __restrict__ l216) {
  const int stride = gridDim.x * blockDim.x;
  const int tid = blockIdx.x * blockDim.x + threadIdx.x;
  for (int i = tid; i < N_H0 * N_INPUTS; i += stride) ft16[i] = (_Float16)ft_w[i];
  for (int i = tid; i < 32 * 512; i += stride)        l116[i] = (_Float16)l1_w[i];
  for (int i = tid; i < 32 * 32; i += stride)         l216[i] = (_Float16)l2_w[i];
}

// ---------------------------------------------------------------------------
// Kernel 1: feature transformer GEMM.
//  Stacked rows: 0..8191 = white, 8192..16383 = black. Per wave: 32 rows x
//  64 cols, K = 10240 in steps of 32. Writes clipped o0 f16 [8192 x 512]
//  with side-dependent concat offset.
// ---------------------------------------------------------------------------
__global__ __launch_bounds__(256) void nnue_ft_gemm(
    const float* __restrict__ wf, const float* __restrict__ bfeat,
    const float* __restrict__ ft_b, const int* __restrict__ side,
    const _Float16* __restrict__ ft16, _Float16* __restrict__ o0) {
  const int lane = threadIdx.x & 31;
  const int wave = threadIdx.x >> 5;
  const int gw   = blockIdx.x * 8 + wave;   // 0..2047
  const int colTile = gw & 3;               // 4 tiles of 64 cols
  const int rowTile = gw >> 2;              // 512 tiles of 32 stacked rows
  const int l15  = lane & 15;
  const int hi   = lane >> 4;
  const int aoff = hi ? 8 : 0;

  const int rowBase   = rowTile * 32;
  const bool isWhite  = rowBase < BATCH;
  const float* feat   = isWhite ? wf : bfeat;
  const int rAdj      = isWhite ? rowBase : rowBase - BATCH;

  const float* aptr0 = feat + (size_t)(rAdj + l15) * N_INPUTS + aoff;
  const float* aptr1 = aptr0 + (size_t)16 * N_INPUTS;

  const int nBase = colTile * 64;
  const _Float16* bptr[4];
#pragma unroll
  for (int t = 0; t < 4; ++t)
    bptr[t] = ft16 + (size_t)(nBase + t * 16 + l15) * N_INPUTS + aoff;

  v8f acc[2][4];
#pragma unroll
  for (int r = 0; r < 2; ++r)
#pragma unroll
    for (int t = 0; t < 4; ++t) acc[r][t] = (v8f)(0.0f);

  for (int k0 = 0; k0 < N_INPUTS; k0 += 32) {
    // prefetch the streaming feature rows a few lines ahead
    __builtin_prefetch(aptr0 + k0 + 256, 0, 3);
    __builtin_prefetch(aptr1 + k0 + 256, 0, 3);
    v16h a0 = frag_from_f32(aptr0 + k0);
    v16h a1 = frag_from_f32(aptr1 + k0);
#pragma unroll
    for (int t = 0; t < 4; ++t) {
      v16h bf16 = frag_from_halfs(bptr[t] + k0);
      acc[0][t] = wmma_f16(a0, bf16, acc[0][t]);
      acc[1][t] = wmma_f16(a1, bf16, acc[1][t]);
    }
  }

  const int s    = side[0];
  const int offs = (((s != 0) ? 1 : 0) == (isWhite ? 1 : 0)) ? 0 : N_H0;
#pragma unroll
  for (int t = 0; t < 4; ++t) {
    const int n = nBase + t * 16 + l15;            // output channel (N dim)
    const float bias = ft_b[n];
#pragma unroll
    for (int r = 0; r < 2; ++r) {
#pragma unroll
      for (int v = 0; v < 8; ++v) {
        // C layout: value v in lane -> row (v + 8*hi), col l15 of the tile
        const int brow = rAdj + r * 16 + hi * 8 + v;
        float x = clamp01(acc[r][t][v] + bias);
        o0[(size_t)brow * 512 + offs + n] = (_Float16)x;
      }
    }
  }
}

// ---------------------------------------------------------------------------
// Kernel 2: fused MLP 512 -> 32 -> 32 -> 1, one wave per 16 batch rows.
// ---------------------------------------------------------------------------
__global__ __launch_bounds__(256) void nnue_mlp(
    const _Float16* __restrict__ o0,
    const _Float16* __restrict__ l116, const float* __restrict__ l1_b,
    const _Float16* __restrict__ l216, const float* __restrict__ l2_b,
    const float* __restrict__ l3_w,    const float* __restrict__ l3_b,
    float* __restrict__ out) {
  __shared__ _Float16 sh1[8][16][32];
  __shared__ _Float16 sh2[8][16][32];

  const int lane = threadIdx.x & 31;
  const int wave = threadIdx.x >> 5;
  const int l15  = lane & 15;
  const int hi   = lane >> 4;
  const int aoff = hi ? 8 : 0;
  const int r0   = (blockIdx.x * 8 + wave) * 16;   // batch row base

  // ---- layer 1: [16 x 512] @ [512 x 32] ----
  v8f acc1[2];
  acc1[0] = (v8f)(0.0f);
  acc1[1] = (v8f)(0.0f);
  const _Float16* ap  = o0   + (size_t)(r0 + l15) * 512 + aoff;
  const _Float16* bp0 = l116 + (size_t)(l15)      * 512 + aoff;
  const _Float16* bp1 = l116 + (size_t)(16 + l15) * 512 + aoff;
#pragma unroll
  for (int k0 = 0; k0 < 512; k0 += 32) {
    v16h a = frag_from_halfs(ap + k0);
    acc1[0] = wmma_f16(a, frag_from_halfs(bp0 + k0), acc1[0]);
    acc1[1] = wmma_f16(a, frag_from_halfs(bp1 + k0), acc1[1]);
  }
#pragma unroll
  for (int t = 0; t < 2; ++t) {
    const float bias = l1_b[t * 16 + l15];
#pragma unroll
    for (int v = 0; v < 8; ++v)
      sh1[wave][hi * 8 + v][t * 16 + l15] = (_Float16)clamp01(acc1[t][v] + bias);
  }
  asm volatile("s_wait_dscnt 0" ::: "memory");

  // ---- layer 2: [16 x 32] @ [32 x 32], single K step ----
  v16h a2 = frag_from_halfs(&sh1[wave][l15][aoff]);
  v8f acc2[2];
  acc2[0] = (v8f)(0.0f);
  acc2[1] = (v8f)(0.0f);
  acc2[0] = wmma_f16(a2, frag_from_halfs(l216 + (size_t)(l15) * 32 + aoff),      acc2[0]);
  acc2[1] = wmma_f16(a2, frag_from_halfs(l216 + (size_t)(16 + l15) * 32 + aoff), acc2[1]);
#pragma unroll
  for (int t = 0; t < 2; ++t) {
    const float bias = l2_b[t * 16 + l15];
#pragma unroll
    for (int v = 0; v < 8; ++v)
      sh2[wave][hi * 8 + v][t * 16 + l15] = (_Float16)clamp01(acc2[t][v] + bias);
  }
  asm volatile("s_wait_dscnt 0" ::: "memory");

  // ---- layer 3: 32-wide dot per row, scale by 300 ----
  if (lane < 16) {
    float s = l3_b[0];
#pragma unroll
    for (int k = 0; k < 32; ++k)
      s += (float)sh2[wave][l15][k] * l3_w[k];
    out[r0 + l15] = s * 300.0f;
  }
}

// ---------------------------------------------------------------------------
extern "C" void kernel_launch(void* const* d_in, const int* in_sizes, int n_in,
                              void* d_out, int out_size, void* d_ws, size_t ws_size,
                              hipStream_t stream) {
  const float* wf    = (const float*)d_in[0];   // [8192, 10240]
  const float* bfeat = (const float*)d_in[1];   // [8192, 10240]
  const int*   side  = (const int*)d_in[2];     // scalar
  const float* ft_w  = (const float*)d_in[3];   // [256, 10240]
  const float* ft_b  = (const float*)d_in[4];   // [256]
  const float* l1_w  = (const float*)d_in[5];   // [32, 512]
  const float* l1_b  = (const float*)d_in[6];   // [32]
  const float* l2_w  = (const float*)d_in[7];   // [32, 32]
  const float* l2_b  = (const float*)d_in[8];   // [32]
  const float* l3_w  = (const float*)d_in[9];   // [1, 32]
  const float* l3_b  = (const float*)d_in[10];  // [1]
  float* out = (float*)d_out;                   // [8192]

  // workspace layout (f16): ft_w16 | l1_w16 | l2_w16 | o0   (~13.7 MB total)
  _Float16* ft16 = (_Float16*)d_ws;
  _Float16* l116 = ft16 + (size_t)N_H0 * N_INPUTS;   // 2,621,440
  _Float16* l216 = l116 + 32 * 512;                  // +16,384
  _Float16* o0   = l216 + 32 * 32;                   // +1,024 ; [8192 x 512]

  // 0) weight conversion
  nnue_cvt<<<1024, 256, 0, stream>>>(ft_w, l1_w, l2_w, ft16, l116, l216);

  // 1) feature-transformer GEMM: 2048 waves -> 256 blocks x 256 threads
  nnue_ft_gemm<<<256, 256, 0, stream>>>(wf, bfeat, ft_b, side, ft16, o0);

  // 2) fused MLP: 512 waves -> 64 blocks x 256 threads
  nnue_mlp<<<64, 256, 0, stream>>>(o0, l116, l1_b, l216, l2_b, l3_w, l3_b, out);
}
